// ImprovedMultiHeadAttention_67233418051608
// MI455X (gfx1250) — compile-verified, matrix operand order checked by reading, not run on hardware
//
#include <hip/hip_runtime.h>
#include <hip/hip_fp16.h>

typedef __attribute__((ext_vector_type(16))) _Float16 v16h;
typedef __attribute__((ext_vector_type(8)))  _Float16 v8h;
typedef __attribute__((ext_vector_type(8)))  float    v8f;
typedef __attribute__((ext_vector_type(4)))  float    v4f;

#define WMMA_F16(a, b, c) \
  __builtin_amdgcn_wmma_f32_16x16x32_f16(false, (a), false, (b), (short)0, (c), false, false)

constexpr int Bc = 4, Sc = 2048, Dc = 512, Hc = 8, DKc = 64;

// ---------------------------------------------------------------------------
// Wt[w][n][k] = (f16) W[w][k][n]   (transposed f16 weights so B-operand rows
// are contiguous in K for WMMA per-lane b128 loads)
// ---------------------------------------------------------------------------
__global__ __launch_bounds__(256) void prep_weights(
    const float* __restrict__ Wq, const float* __restrict__ Wk,
    const float* __restrict__ Wv, const float* __restrict__ Wfc,
    _Float16* __restrict__ Wt) {
  int w = blockIdx.y;
  const float* src = (w == 0) ? Wq : (w == 1) ? Wk : (w == 2) ? Wv : Wfc;
  int idx = blockIdx.x * blockDim.x + threadIdx.x;  // 0..512*512-1
  int n = idx >> 9;
  int k = idx & 511;
  Wt[(size_t)w * Dc * Dc + idx] = (_Float16)src[(size_t)k * Dc + n];
}

// ---------------------------------------------------------------------------
// X (M=8192 x K=512, f32) @ Wt^T -> head-split f16 output.
// transposed==0: out[((b*H+h)*S + s)*64 + dk]   (q/k heads, row-major [S,64])
// transposed==1: out[((b*H+h)*64 + dk)*S + s]   (v heads, [64,S] for AV B-op)
// One wave per 16x16 tile, 8 waves/block over N.
// ---------------------------------------------------------------------------
__global__ __launch_bounds__(256) void proj_gemm(
    const float* __restrict__ X, const _Float16* __restrict__ Wt,
    _Float16* __restrict__ out, int transposed) {
  int wave = threadIdx.x >> 5;
  int lane = threadIdx.x & 31;
  int mt = blockIdx.x;                 // 0..511
  int nt = blockIdx.y * 8 + wave;      // 0..31
  int mlane = lane & 15;
  int khi = (lane >> 4) << 3;          // 0 or 8: lane's K sub-block

  const float*    arow = X  + (size_t)(mt * 16 + mlane) * Dc;
  const _Float16* brow = Wt + (size_t)(nt * 16 + mlane) * Dc;

  v8f c = {};
  for (int kb = 0; kb < Dc; kb += 32) {
    v4f x0 = *(const v4f*)(arow + kb + khi);
    v4f x1 = *(const v4f*)(arow + kb + khi + 4);
    v4f x2 = *(const v4f*)(arow + kb + khi + 16);
    v4f x3 = *(const v4f*)(arow + kb + khi + 20);
    v16h a;
    for (int i = 0; i < 4; ++i) {
      a[i]      = (_Float16)x0[i];
      a[4 + i]  = (_Float16)x1[i];
      a[8 + i]  = (_Float16)x2[i];
      a[12 + i] = (_Float16)x3[i];
    }
    v8h b0 = *(const v8h*)(brow + kb + khi);
    v8h b1 = *(const v8h*)(brow + kb + khi + 16);
    v16h b;
    for (int i = 0; i < 8; ++i) { b[i] = b0[i]; b[8 + i] = b1[i]; }
    c = WMMA_F16(a, b, c);
  }

  int nglob = nt * 16 + mlane;
  int h = nglob >> 6, dk = nglob & 63;
  for (int i = 0; i < 8; ++i) {
    int mg = mt * 16 + i + ((lane >> 4) << 3);
    int b = mg >> 11;           // / 2048
    int s = mg & (Sc - 1);
    _Float16 val = (_Float16)c[i];
    size_t bh = (size_t)b * Hc + h;
    if (!transposed) out[(bh * Sc + s) * DKc + dk] = val;
    else             out[(bh * DKc + dk) * Sc + s] = val;
  }
}

// ---------------------------------------------------------------------------
// Fused attention per (b,h,q-tile16):
//   1) scores = Q K^T * scale + rel_bias, masked  -> LDS (16x2048 f32)
//   2) row softmax stats in LDS
//   3) attn written once with non-temporal stores (537 MB stream, skip L2)
//   4) ctx = P @ V computed from the exp values still in LDS (no HBM re-read
//      of attn): 8 waves x 256-key stripes -> partials -> LDS tree reduce,
//      rows scaled by 1/rowsum, stored f16.
// Dynamic LDS: 16*2048 scores + 16*16 red + 16 rowsums (~132 KB).
// ---------------------------------------------------------------------------
__global__ __launch_bounds__(256) void attn_fused(
    const _Float16* __restrict__ qh, const _Float16* __restrict__ kh,
    const _Float16* __restrict__ vt, const int* __restrict__ mask,
    const float* __restrict__ rel_bias, float* __restrict__ attn,
    _Float16* __restrict__ ctx) {
  extern __shared__ float smem[];
  float* sc  = smem;                       // [16][2048] scores / exp / partials
  float* red = smem + 16 * Sc;             // [16][16]
  float* rs  = red + 16 * 16;              // [16] row sums

  int bh = blockIdx.y;                     // 0..31
  int b  = bh >> 3;
  int h  = bh & 7;
  int mt = blockIdx.x;                     // 0..127
  int wave = threadIdx.x >> 5;
  int lane = threadIdx.x & 31;
  int mlane = lane & 15;
  int khi = (lane >> 4) << 3;

  // ---- Phase 1: scores -> LDS -------------------------------------------
  const _Float16* qrow = qh + ((size_t)bh * Sc + mt * 16 + mlane) * DKc;
  v16h a0, a1;
  {
    v8h t0 = *(const v8h*)(qrow + khi);
    v8h t1 = *(const v8h*)(qrow + khi + 16);
    v8h t2 = *(const v8h*)(qrow + 32 + khi);
    v8h t3 = *(const v8h*)(qrow + 32 + khi + 16);
    for (int i = 0; i < 8; ++i) {
      a0[i] = t0[i]; a0[8 + i] = t1[i];
      a1[i] = t2[i]; a1[8 + i] = t3[i];
    }
  }

  const float scale = 0.125f;  // 64^-0.5
  for (int kt = 0; kt < 16; ++kt) {
    int kb = wave * 256 + kt * 16;
    const _Float16* krow = kh + ((size_t)bh * Sc + kb + mlane) * DKc;
    v8h t0 = *(const v8h*)(krow + khi);
    v8h t1 = *(const v8h*)(krow + khi + 16);
    v8h t2 = *(const v8h*)(krow + 32 + khi);
    v8h t3 = *(const v8h*)(krow + 32 + khi + 16);
    v16h b0, b1;
    for (int i = 0; i < 8; ++i) {
      b0[i] = t0[i]; b0[8 + i] = t1[i];
      b1[i] = t2[i]; b1[8 + i] = t3[i];
    }
    v8f c = {};
    c = WMMA_F16(a0, b0, c);
    c = WMMA_F16(a1, b1, c);

    int key = kb + mlane;
    int km = mask[(size_t)b * Sc + key];
    for (int i = 0; i < 8; ++i) {
      int mloc = i + ((lane >> 4) << 3);
      int qpos = mt * 16 + mloc;
      int d = qpos - key;                       // context - memory
      int bucket = d < 0 ? 0 : (d > 32 ? 32 : d);
      float val = (km == 0) ? -1e9f
                            : (c[i] * scale + rel_bias[bucket * Hc + h]);
      sc[mloc * Sc + key] = val;
    }
  }
  __syncthreads();

  // ---- Phase 2: softmax stats (16 threads per row) ----------------------
  int g = threadIdx.x >> 4;   // row 0..15
  int j = threadIdx.x & 15;
  float mx = -3.0e38f;
  for (int k = j; k < Sc; k += 16) mx = fmaxf(mx, sc[g * Sc + k]);
  red[g * 16 + j] = mx;
  __syncthreads();
  mx = red[g * 16];
  for (int i = 1; i < 16; ++i) mx = fmaxf(mx, red[g * 16 + i]);

  float sum = 0.f;
  for (int k = j; k < Sc; k += 16) {
    float e = __expf(sc[g * Sc + k] - mx);
    sc[g * Sc + k] = e;                         // keep exp values in LDS
    sum += e;
  }
  __syncthreads();             // finish reads of red before reuse
  red[g * 16 + j] = sum;
  __syncthreads();
  sum = 0.f;
  for (int i = 0; i < 16; ++i) sum += red[g * 16 + i];
  if (j == 0) rs[g] = sum;
  float inv = 1.0f / sum;

  // ---- Phase 3: stream normalized attn out (non-temporal) ---------------
  float* orow = attn + ((size_t)bh * Sc + mt * 16 + g) * Sc;
  for (int k = j; k < Sc; k += 16)
    __builtin_nontemporal_store(sc[g * Sc + k] * inv, &orow[k]);

  // ---- Phase 4: ctx = P @ V from LDS exp values -------------------------
  // Each wave covers keys [wave*256, wave*256+256); 4 n-tiles of d_v.
  v8f acc[4] = {{}, {}, {}, {}};
  const _Float16* vbase = vt + (size_t)bh * DKc * Sc;   // [64][S] f16
  for (int kk = 0; kk < 256; kk += 32) {
    int kb = wave * 256 + kk;
    const float* ar = sc + (size_t)mlane * Sc + kb + khi;
    v4f x0 = *(const v4f*)(ar);
    v4f x1 = *(const v4f*)(ar + 4);
    v4f x2 = *(const v4f*)(ar + 16);
    v4f x3 = *(const v4f*)(ar + 20);
    v16h a;
    for (int i = 0; i < 4; ++i) {
      a[i]      = (_Float16)x0[i];
      a[4 + i]  = (_Float16)x1[i];
      a[8 + i]  = (_Float16)x2[i];
      a[12 + i] = (_Float16)x3[i];
    }
    for (int nt = 0; nt < 4; ++nt) {
      const _Float16* br = vbase + (size_t)(nt * 16 + mlane) * Sc + kb + khi;
      v8h b0 = *(const v8h*)(br);
      v8h b1 = *(const v8h*)(br + 16);
      v16h bb;
      for (int i = 0; i < 8; ++i) { bb[i] = b0[i]; bb[8 + i] = b1[i]; }
      acc[nt] = WMMA_F16(a, bb, acc[nt]);
    }
  }
  __syncthreads();             // all waves done reading sc; safe to overwrite

  // partials: pbuf[wave][nt][16][16] = 32 KB, reusing the score region
  for (int nt = 0; nt < 4; ++nt) {
    for (int i = 0; i < 8; ++i) {
      int M = i + ((lane >> 4) << 3);
      sc[(((wave * 4 + nt) * 16 + M) << 4) + mlane] = acc[nt][i];
    }
  }
  __syncthreads();

  // tree reduce over 8 waves; 256 threads x 4 outputs = 16x64
  int e0 = threadIdx.x * 4;
  for (int e = e0; e < e0 + 4; ++e) {
    int m  = e >> 6;
    int d2 = e & 63;
    int nt = d2 >> 4, n = d2 & 15;
    float s2 = 0.f;
    for (int w = 0; w < 8; ++w) s2 += sc[(((w * 4 + nt) * 16 + m) << 4) + n];
    float invm = 1.0f / rs[m];
    int s_tok = mt * 16 + m;
    ctx[((size_t)b * Sc + s_tok) * (Hc * DKc) + h * DKc + d2] =
        (_Float16)(s2 * invm);
  }
}

// ---------------------------------------------------------------------------
// out = ctx (8192x512 f16) @ Wfc  -> f32
// ---------------------------------------------------------------------------
__global__ __launch_bounds__(256) void final_gemm(
    const _Float16* __restrict__ ctx, const _Float16* __restrict__ Wt,
    float* __restrict__ out) {
  int wave = threadIdx.x >> 5;
  int lane = threadIdx.x & 31;
  int mt = blockIdx.x;               // 0..511
  int nt = blockIdx.y * 8 + wave;    // 0..31
  int mlane = lane & 15;
  int khi = (lane >> 4) << 3;

  const _Float16* arow = ctx + (size_t)(mt * 16 + mlane) * Dc;
  const _Float16* brow = Wt + (size_t)(nt * 16 + mlane) * Dc;

  v8f c = {};
  for (int kb = 0; kb < Dc; kb += 32) {
    v8h t0 = *(const v8h*)(arow + kb + khi);
    v8h t1 = *(const v8h*)(arow + kb + khi + 16);
    v8h u0 = *(const v8h*)(brow + kb + khi);
    v8h u1 = *(const v8h*)(brow + kb + khi + 16);
    v16h a, b;
    for (int i = 0; i < 8; ++i) {
      a[i] = t0[i]; a[8 + i] = t1[i];
      b[i] = u0[i]; b[8 + i] = u1[i];
    }
    c = WMMA_F16(a, b, c);
  }

  int n = nt * 16 + mlane;
  for (int i = 0; i < 8; ++i) {
    int m = mt * 16 + i + ((lane >> 4) << 3);
    out[(size_t)m * Dc + n] = c[i];
  }
}

// ---------------------------------------------------------------------------
extern "C" void kernel_launch(void* const* d_in, const int* in_sizes, int n_in,
                              void* d_out, int out_size, void* d_ws, size_t ws_size,
                              hipStream_t stream) {
  const float* q        = (const float*)d_in[0];
  const float* k        = (const float*)d_in[1];
  const float* v        = (const float*)d_in[2];
  const int*   mask     = (const int*)d_in[3];
  const float* Wq       = (const float*)d_in[4];
  const float* Wk       = (const float*)d_in[5];
  const float* Wv       = (const float*)d_in[6];
  const float* Wfc      = (const float*)d_in[7];
  const float* rel_bias = (const float*)d_in[8];

  float* out  = (float*)d_out;
  float* attn = out + (size_t)Bc * Sc * Dc;   // second tuple element

  const size_t headElems = (size_t)Bc * Hc * Sc * DKc;  // 4,194,304
  char* ws = (char*)d_ws;
  _Float16* Wt  = (_Float16*)ws;                        // 4 * 512*512 f16 (2 MB)
  _Float16* qh  = (_Float16*)(ws + (size_t)(2u << 20)); // 8 MB
  _Float16* kh  = qh + headElems;                       // 8 MB
  _Float16* vt  = kh + headElems;                       // 8 MB  ([B,H,64,S])
  _Float16* ctx = vt + headElems;                       // 8 MB  ([B,S,512])

  prep_weights<<<dim3(1024, 4), 256, 0, stream>>>(Wq, Wk, Wv, Wfc, Wt);

  proj_gemm<<<dim3(512, 4), 256, 0, stream>>>(q, Wt + 0 * (size_t)Dc * Dc, qh, 0);
  proj_gemm<<<dim3(512, 4), 256, 0, stream>>>(k, Wt + 1 * (size_t)Dc * Dc, kh, 0);
  proj_gemm<<<dim3(512, 4), 256, 0, stream>>>(v, Wt + 2 * (size_t)Dc * Dc, vt, 1);

  size_t smemBytes = (size_t)(16 * Sc + 16 * 16 + 16) * sizeof(float); // ~132 KB
  attn_fused<<<dim3(128, 32), 256, smemBytes, stream>>>(
      qh, kh, vt, mask, rel_bias, attn, ctx);

  final_gemm<<<dim3(512, 4), 256, 0, stream>>>(ctx, Wt + 3 * (size_t)Dc * Dc, out);
}
